// USDN_32289564131395
// MI455X (gfx1250) — matrix operand order repaired and checked by reading, not compiled, not run on hardware
//
#include <hip/hip_runtime.h>
#include <hip/hip_bf16.h>
#include <math.h>

typedef _Float16 v16h __attribute__((ext_vector_type(16)));
typedef _Float16 v8h  __attribute__((ext_vector_type(8)));
typedef float    v8f  __attribute__((ext_vector_type(8)));

// ---------------------------------------------------------------------------
// Prep: convert an f32 weight matrix [Ksrc x Nsrc] (row-major) into f16
// WMMA B-fragment-major layout, zero-padded to K = KT*32, N = NT*16.
// Fragment (kc,nc) = 512 f16: lane l holds column (nc*16 + (l&15)),
// K {0..7,16..23} (low lanes) / {8..15,24..31} (high lanes) of kc*32, 16 f16 per lane.
// ---------------------------------------------------------------------------
__global__ void usdn_prep_weights(const float* __restrict__ src,
                                  _Float16* __restrict__ dst,
                                  int Ksrc, int Nsrc, int NT, int total) {
    int e = blockIdx.x * blockDim.x + threadIdx.x;
    if (e >= total) return;
    int i    = e & 15;
    int lane = (e >> 4) & 31;
    int frag = e >> 9;
    int nc = frag % NT;
    int kc = frag / NT;
    int k = kc * 32 + (i < 8 ? ((lane >> 4) << 3) + i : 16 + ((lane >> 4) << 3) + (i - 8));
    int n = nc * 16 + (lane & 15);
    float v = (k < Ksrc && n < Nsrc) ? src[(size_t)k * Nsrc + n] : 0.0f;
    dst[e] = (_Float16)v;
}

// ---------------------------------------------------------------------------
// Fused forward: each block = 32 rows, 8 waves (wave32).
//   sA: Yh tile (32x192 f16) -> later h2 (32x256 f16)
//   sB: h1 (32x256 f16)      -> later d  (32x256 f16)
//   sAl/sBe: softplus heads (32x32 f32); sPi: stick weights (32x32 f16)
// ---------------------------------------------------------------------------
__global__ __launch_bounds__(256)
void usdn_fused_mlp(const float* __restrict__ Yh, const float* __restrict__ u,
                    const float* __restrict__ b1, const float* __restrict__ b2,
                    const float* __restrict__ ba, const float* __restrict__ bb,
                    const float* __restrict__ bd1, const float* __restrict__ bd2,
                    const _Float16* __restrict__ W1f, const _Float16* __restrict__ W2f,
                    const _Float16* __restrict__ Waf, const _Float16* __restrict__ Wbf,
                    const _Float16* __restrict__ Wd1f, const _Float16* __restrict__ Wd2f,
                    float* __restrict__ outRecon, float* __restrict__ outAlpha,
                    float* __restrict__ outBeta,  float* __restrict__ outPi,
                    int N)
{
    __shared__ __align__(16) _Float16 sA[32 * 256];
    __shared__ __align__(16) _Float16 sB[32 * 256];
    __shared__ __align__(16) float    sAl[32 * 32];
    __shared__ __align__(16) float    sBe[32 * 32];
    __shared__ __align__(16) _Float16 sPi[32 * 32];

    const int tid  = threadIdx.x;
    const int lane = tid & 31;
    const int w    = tid >> 5;
    const int sel  = lane >> 4;   // K-half selector for A/C layouts
    const int nl   = lane & 15;   // N-column within tile
    const int rowbase = blockIdx.x * 32;
    if (rowbase >= N) return;

    // A-fragment gather from LDS (16-bit A 16x32 layout):
    // low lanes hold K {0..7,16..23}, high lanes {8..15,24..31} of row (lane&15).
    auto loadA = [&](const _Float16* s, int ld, int row, int kbase) -> v16h {
        const v8h* p0 = (const v8h*)(s + row * ld + kbase + sel * 8);
        const v8h* p1 = (const v8h*)(s + row * ld + kbase + 16 + sel * 8);
        v8h lo = *p0, hi = *p1;
        v16h a;
#pragma unroll
        for (int i = 0; i < 8; ++i) { a[i] = lo[i]; a[i + 8] = hi[i]; }
        return a;
    };

    // ---- Stage Yh tile: 32 x 162 f32 -> 32 x 192 f16 (zero padded) ----
    for (int idx = tid; idx < 32 * 192; idx += 256) {
        int r = idx / 192, c = idx % 192;
        float v = (c < 162) ? Yh[(size_t)(rowbase + r) * 162 + c] : 0.0f;
        sA[r * 192 + c] = (_Float16)v;
    }
    __syncthreads();

    // ---- Layer 1: h1 = relu(Yh @ W1 + b1)  [K=192/6 steps, Nout=256] ----
    {
        const int mt  = w >> 2;
        const int nt0 = (w & 3) * 4;
        v8f acc[4] = {};
#pragma unroll
        for (int kc = 0; kc < 6; ++kc) {
            v16h a = loadA(sA, 192, mt * 16 + nl, kc * 32);
#pragma unroll
            for (int j = 0; j < 4; ++j) {
                v16h bm = ((const v16h*)(W1f + (size_t)(kc * 16 + nt0 + j) * 512))[lane];
                acc[j] = __builtin_amdgcn_wmma_f32_16x16x32_f16(
                    false, a, false, bm, (short)0, acc[j], false, false);
            }
        }
#pragma unroll
        for (int j = 0; j < 4; ++j) {
            int n = (nt0 + j) * 16 + nl;
            float bias = b1[n];
#pragma unroll
            for (int r = 0; r < 8; ++r) {
                int m = mt * 16 + sel * 8 + r;
                float v = acc[j][r] + bias;
                sB[m * 256 + n] = (_Float16)(v > 0.0f ? v : 0.0f);
            }
        }
    }
    __syncthreads();

    // ---- Layer 2: h2 = relu(h1 @ W2 + b2)  [K=256/8 steps, Nout=256] ----
    {
        const int mt  = w >> 2;
        const int nt0 = (w & 3) * 4;
        v8f acc[4] = {};
#pragma unroll
        for (int kc = 0; kc < 8; ++kc) {
            v16h a = loadA(sB, 256, mt * 16 + nl, kc * 32);
#pragma unroll
            for (int j = 0; j < 4; ++j) {
                v16h bm = ((const v16h*)(W2f + (size_t)(kc * 16 + nt0 + j) * 512))[lane];
                acc[j] = __builtin_amdgcn_wmma_f32_16x16x32_f16(
                    false, a, false, bm, (short)0, acc[j], false, false);
            }
        }
#pragma unroll
        for (int j = 0; j < 4; ++j) {
            int n = (nt0 + j) * 16 + nl;
            float bias = b2[n];
#pragma unroll
            for (int r = 0; r < 8; ++r) {
                int m = mt * 16 + sel * 8 + r;
                float v = acc[j][r] + bias;
                sA[m * 256 + n] = (_Float16)(v > 0.0f ? v : 0.0f);
            }
        }
    }
    __syncthreads();

    // ---- Heads: alpha/beta = softplus(h2 @ W{a,b} + b{a,b}) + EPS  [Nout=32pad] ----
    {
        const int mt     = w >> 2;
        const int isBeta = (w >> 1) & 1;
        const int nt     = w & 1;
        const _Float16* wf = isBeta ? Wbf : Waf;
        v8f acc = {};
#pragma unroll
        for (int kc = 0; kc < 8; ++kc) {
            v16h a  = loadA(sA, 256, mt * 16 + nl, kc * 32);
            v16h bm = ((const v16h*)(wf + (size_t)(kc * 2 + nt) * 512))[lane];
            acc = __builtin_amdgcn_wmma_f32_16x16x32_f16(
                false, a, false, bm, (short)0, acc, false, false);
        }
        int n = nt * 16 + nl;
        const float* bp = isBeta ? bb : ba;
        float bias = (n < 30) ? bp[n] : 0.0f;
        float* sH  = isBeta ? sBe : sAl;
        float* oH  = isBeta ? outBeta : outAlpha;
#pragma unroll
        for (int r = 0; r < 8; ++r) {
            int m = mt * 16 + sel * 8 + r;
            float x  = acc[r] + bias;
            float sp = (x > 20.0f) ? x : log1pf(expf(x));
            sp += 1e-4f;
            sH[m * 32 + n] = sp;
            if (n < 30) oH[(size_t)(rowbase + m) * 30 + n] = sp;
        }
    }
    __syncthreads();

    // ---- Stick-breaking (sequential scan, lane = local row on wave 0) ----
    if (w == 0) {
        int m = lane;
        float uu  = u[0] * 0.98f + 0.01f;
        float l2u = log2f(uu);
        float cp  = 1.0f;
        for (int k = 0; k < 30; ++k) {
            float vv;
            if (k == 29) {
                vv = 1.0f;
            } else {
                float a = sAl[m * 32 + k];
                float b = sBe[m * 32 + k];
                float t = 1.0f - exp2f(l2u / b);     // 1 - uu^(1/beta)
                t = fmaxf(t, 1e-30f);
                vv = exp2f(log2f(t) / a);            // t^(1/alpha)
            }
            float pi = vv * cp;
            sPi[m * 32 + k] = (_Float16)pi;
            outPi[(size_t)(rowbase + m) * 30 + k] = pi;
            cp *= (1.0f - vv);
        }
        sPi[m * 32 + 30] = (_Float16)0.0f;
        sPi[m * 32 + 31] = (_Float16)0.0f;
    }
    __syncthreads();

    // ---- Decoder 1: d = relu(pi @ Wd1 + bd1)  [K=32/1 step, Nout=256] ----
    {
        const int mt  = w >> 2;
        const int nt0 = (w & 3) * 4;
        v8f acc[4] = {};
        v16h a = loadA(sPi, 32, mt * 16 + nl, 0);
#pragma unroll
        for (int j = 0; j < 4; ++j) {
            v16h bm = ((const v16h*)(Wd1f + (size_t)(nt0 + j) * 512))[lane];
            acc[j] = __builtin_amdgcn_wmma_f32_16x16x32_f16(
                false, a, false, bm, (short)0, acc[j], false, false);
        }
#pragma unroll
        for (int j = 0; j < 4; ++j) {
            int n = (nt0 + j) * 16 + nl;
            float bias = bd1[n];
#pragma unroll
            for (int r = 0; r < 8; ++r) {
                int m = mt * 16 + sel * 8 + r;
                float v = acc[j][r] + bias;
                sB[m * 256 + n] = (_Float16)(v > 0.0f ? v : 0.0f);
            }
        }
    }
    __syncthreads();

    // ---- Decoder 2: recon = sigmoid(d @ Wd2 + bd2)  [K=256/8, Nout=192pad] ----
    {
#pragma unroll
        for (int j = 0; j < 3; ++j) {
            int t  = w * 3 + j;          // 24 tiles total: 2 M-tiles x 12 N-tiles
            int mt = t / 12;
            int nt = t % 12;
            v8f acc = {};
#pragma unroll
            for (int kc = 0; kc < 8; ++kc) {
                v16h a  = loadA(sB, 256, mt * 16 + nl, kc * 32);
                v16h bm = ((const v16h*)(Wd2f + (size_t)(kc * 12 + nt) * 512))[lane];
                acc = __builtin_amdgcn_wmma_f32_16x16x32_f16(
                    false, a, false, bm, (short)0, acc, false, false);
            }
            int n = nt * 16 + nl;
            float bias = (n < 162) ? bd2[n] : 0.0f;
#pragma unroll
            for (int r = 0; r < 8; ++r) {
                int m = mt * 16 + sel * 8 + r;
                float x = acc[r] + bias;
                float y = 1.0f / (1.0f + expf(-x));
                if (n < 162) outRecon[(size_t)(rowbase + m) * 162 + n] = y;
            }
        }
    }
}

extern "C" void kernel_launch(void* const* d_in, const int* in_sizes, int n_in,
                              void* d_out, int out_size, void* d_ws, size_t ws_size,
                              hipStream_t stream) {
    const float* Yh  = (const float*)d_in[0];
    const float* u   = (const float*)d_in[1];
    const float* W1  = (const float*)d_in[2];
    const float* b1  = (const float*)d_in[3];
    const float* W2  = (const float*)d_in[4];
    const float* b2  = (const float*)d_in[5];
    const float* Wa  = (const float*)d_in[6];
    const float* ba  = (const float*)d_in[7];
    const float* Wb  = (const float*)d_in[8];
    const float* bb  = (const float*)d_in[9];
    const float* Wd1 = (const float*)d_in[10];
    const float* bd1 = (const float*)d_in[11];
    const float* Wd2 = (const float*)d_in[12];
    const float* bd2 = (const float*)d_in[13];

    const int N = in_sizes[0] / 162;

    // f16 fragment-major weights in workspace (~368 KB total)
    _Float16* W1f  = (_Float16*)d_ws;        // 6*16*512  = 49152
    _Float16* W2f  = W1f  + 49152;           // 8*16*512  = 65536
    _Float16* Waf  = W2f  + 65536;           // 8*2*512   = 8192
    _Float16* Wbf  = Waf  + 8192;            // 8192
    _Float16* Wd1f = Wbf  + 8192;            // 1*16*512  = 8192
    _Float16* Wd2f = Wd1f + 8192;            // 8*12*512  = 49152

    auto prep = [&](const float* src, _Float16* dst, int K, int Nn, int NT, int KT) {
        int total = KT * NT * 512;
        usdn_prep_weights<<<(total + 255) / 256, 256, 0, stream>>>(src, dst, K, Nn, NT, total);
    };
    prep(W1,  W1f,  162, 256, 16, 6);
    prep(W2,  W2f,  256, 256, 16, 8);
    prep(Wa,  Waf,  256,  30,  2, 8);
    prep(Wb,  Wbf,  256,  30,  2, 8);
    prep(Wd1, Wd1f,  30, 256, 16, 1);
    prep(Wd2, Wd2f, 256, 162, 12, 8);

    float* out      = (float*)d_out;
    float* outRecon = out;
    float* outAlpha = out + (size_t)N * 162;
    float* outBeta  = outAlpha + (size_t)N * 30;
    float* outPi    = outBeta  + (size_t)N * 30;

    usdn_fused_mlp<<<(N + 31) / 32, 256, 0, stream>>>(
        Yh, u, b1, b2, ba, bb, bd1, bd2,
        W1f, W2f, Waf, Wbf, Wd1f, Wd2f,
        outRecon, outAlpha, outBeta, outPi, N);
}
